// SoftTree_44032004718991
// MI455X (gfx1250) — matrix operand order repaired and checked by reading, not compiled, not run on hardware
//
#include <hip/hip_runtime.h>
#include <hip/hip_bf16.h>

#define BATCH    16384
#define IN_F     512
#define OUT_F    64
#define GATES    255
#define LEAVES   256
#define TREE_D   8

#define KC       32     // K chunk per WMMA (bf16)
#define KP       40     // padded K stride in LDS (80B rows, 16B-aligned chunks)
#define BM       256    // batch rows per workgroup (projection kernel)
#define LT       4      // leaves per workgroup (projection kernel)
#define NTHREADS 512    // 16 waves
#define BMG      16     // batch rows per workgroup (gate kernel)

typedef __bf16 v16bf __attribute__((ext_vector_type(16)));
typedef float  v8f   __attribute__((ext_vector_type(8)));

union Frag16 { v16bf v; int4 i[2]; };

__device__ __forceinline__ __bf16 to_bf16(float f) { return (__bf16)f; }

__device__ __forceinline__ float fast_sigmoid(float v) {
    return 1.0f / (1.0f + __expf(-v));
}

// ---------------------------------------------------------------------------
// Kernel 1: out[l,b,o] = act( x[b,:] . pw[l,:,o] + pb[l,o] )
//   act = pi*tanh for o < 32, sigmoid for o >= 32
// grid: (BATCH/BM, LEAVES/LT) = (64,64), block: 512 threads (16 waves)
// Wave w: rows [w*16, w*16+16), all 4 N-tiles, all 4 leaves -> 16 WMMA/kstep
// B LDS layout is transposed [leaf][o][k] with a 16B-chunk XOR swizzle:
//   chunk' = (k>>3) ^ ((o>>4)&3)   (kills the o0-stride bank conflicts;
//   on the fragment-load side the key is just `nt`, compile time)
// ---------------------------------------------------------------------------
__global__ __launch_bounds__(NTHREADS)
void softtree_proj_kernel(const float* __restrict__ x,
                          const float* __restrict__ pw,
                          const float* __restrict__ pb,
                          float* __restrict__ out) {
    __shared__ __align__(16) __bf16 As[BM * KP];            // 20 KB, [row][k]
    __shared__ __align__(16) __bf16 Bs[LT][OUT_F * KP];     // 20 KB, [leaf][o][k] swizzled

    const int tid   = threadIdx.x;
    const int wave  = tid >> 5;
    const int lane  = tid & 31;
    const int laneM = lane & 15;
    const int half  = lane >> 4;

    const int bRow0 = blockIdx.x * BM;
    const int leaf0 = blockIdx.y * LT;

    v8f acc[LT][4] = {};

    for (int k0 = 0; k0 < IN_F; k0 += KC) {
        __syncthreads();  // previous iteration's fragment reads done

        // ---- load A tile 256x32 f32 -> bf16 LDS; thread t: row t>>1, k half (t&1)*16
        {
            const int r   = tid >> 1;
            const int kk0 = (tid & 1) * 16;
            const float* src = x + (size_t)(bRow0 + r) * IN_F + k0 + kk0;
            float4 f0 = *(const float4*)(src + 0);
            float4 f1 = *(const float4*)(src + 4);
            float4 f2 = *(const float4*)(src + 8);
            float4 f3 = *(const float4*)(src + 12);
            __bf16* dst = &As[r * KP + kk0];
            dst[0]  = to_bf16(f0.x); dst[1]  = to_bf16(f0.y);
            dst[2]  = to_bf16(f0.z); dst[3]  = to_bf16(f0.w);
            dst[4]  = to_bf16(f1.x); dst[5]  = to_bf16(f1.y);
            dst[6]  = to_bf16(f1.z); dst[7]  = to_bf16(f1.w);
            dst[8]  = to_bf16(f2.x); dst[9]  = to_bf16(f2.y);
            dst[10] = to_bf16(f2.z); dst[11] = to_bf16(f2.w);
            dst[12] = to_bf16(f3.x); dst[13] = to_bf16(f3.y);
            dst[14] = to_bf16(f3.z); dst[15] = to_bf16(f3.w);
        }

        // ---- load B tile (4 leaves) 32x64 f32, transpose -> Bs[lt][o][k] (swizzled)
        {
            const int lt = tid >> 7;        // 0..3
            const int tt = tid & 127;
            const int kk = tt >> 2;         // 0..31
            const int o0 = (tt & 3) * 16;   // 0,16,32,48
            const float* src = pw + ((size_t)(leaf0 + lt) * IN_F + (k0 + kk)) * OUT_F + o0;
            float4 f0 = *(const float4*)(src + 0);
            float4 f1 = *(const float4*)(src + 4);
            float4 f2 = *(const float4*)(src + 8);
            float4 f3 = *(const float4*)(src + 12);
            float v[16] = {f0.x,f0.y,f0.z,f0.w, f1.x,f1.y,f1.z,f1.w,
                           f2.x,f2.y,f2.z,f2.w, f3.x,f3.y,f3.z,f3.w};
            // swizzle key (o>>4)&3 == tt&3 for all j<16; constant per thread
            const int swz = ((kk >> 3) ^ (tt & 3)) * 8 + (kk & 7);
            #pragma unroll
            for (int j = 0; j < 16; ++j)
                Bs[lt][(o0 + j) * KP + swz] = to_bf16(v[j]);
        }

        __syncthreads();

        // ---- A fragment: lane M = laneM, K chunks [8*half,+8) and [16+8*half,+8)
        Frag16 a;
        {
            const int r  = wave * 16 + laneM;
            const int kb = half * 8;
            a.i[0] = *(const int4*)&As[r * KP + kb];
            a.i[1] = *(const int4*)&As[r * KP + 16 + kb];
        }

        #pragma unroll
        for (int lt = 0; lt < LT; ++lt) {
            #pragma unroll
            for (int nt = 0; nt < 4; ++nt) {
                Frag16 b;
                const int o = nt * 16 + laneM;
                // chunk indices (half + 2*c) ^ nt  (nt is the swizzle key here)
                b.i[0] = *(const int4*)&Bs[lt][o * KP + ((half + 0) ^ nt) * 8];
                b.i[1] = *(const int4*)&Bs[lt][o * KP + ((half + 2) ^ nt) * 8];
                acc[lt][nt] = __builtin_amdgcn_wmma_f32_16x16x32_bf16(
                    false, a.v, false, b.v, (short)0, acc[lt][nt], false, false);
            }
        }
    }

    // ---- epilogue: bias + activation + store
    const float PI_F = 3.14159265358979323846f;
    #pragma unroll
    for (int lt = 0; lt < LT; ++lt) {
        const int leaf = leaf0 + lt;
        #pragma unroll
        for (int nt = 0; nt < 4; ++nt) {
            const int col  = nt * 16 + laneM;
            const float bias = pb[leaf * OUT_F + col];
            #pragma unroll
            for (int r = 0; r < 8; ++r) {
                const int row = bRow0 + wave * 16 + half * 8 + r;
                const float v = acc[lt][nt][r] + bias;
                float res;
                if (nt < 2) {
                    // pi * tanh(v), overflow-safe
                    const float e = __expf(2.0f * v);
                    res = PI_F * (1.0f - 2.0f / (e + 1.0f));
                } else {
                    res = fast_sigmoid(v);
                }
                out[((size_t)leaf * BATCH + row) * OUT_F + col] = res;
            }
        }
    }
}

// ---------------------------------------------------------------------------
// Kernel 2: gates g = sigmoid(x @ gw + gb), then depth-8 product tree ->
//           leaf_probs[l, b].  WG = 256 threads (8 waves), 16 batch rows,
//           256 gate columns (col 255 padded, never read by the tree).
// ---------------------------------------------------------------------------
__global__ __launch_bounds__(256)
void softtree_gate_kernel(const float* __restrict__ x,
                          const float* __restrict__ gw,
                          const float* __restrict__ gb,
                          float* __restrict__ leaf_out) {
    __shared__ __align__(16) __bf16 Ag[BMG * KP];        // 1.25 KB
    __shared__ __align__(16) __bf16 Bg[256 * KP];        // 20 KB  [n][k]
    __shared__ float gbuf[BMG * 256];                    // 16 KB  [row][col]

    const int tid   = threadIdx.x;
    const int wave  = tid >> 5;
    const int lane  = tid & 31;
    const int laneM = lane & 15;
    const int half  = lane >> 4;

    const int bRow0 = blockIdx.x * BMG;

    v8f acc[2] = {};

    for (int k0 = 0; k0 < IN_F; k0 += KC) {
        __syncthreads();

        // A tile 16x32: first 128 threads, float4 each
        if (tid < 128) {
            const int r  = tid >> 3;
            const int kk = (tid & 7) * 4;
            float4 f = *(const float4*)(x + (size_t)(bRow0 + r) * IN_F + k0 + kk);
            __bf16* dst = &Ag[r * KP + kk];
            dst[0] = to_bf16(f.x); dst[1] = to_bf16(f.y);
            dst[2] = to_bf16(f.z); dst[3] = to_bf16(f.w);
        }

        // B tile 32x256 (transpose, pad col >= 255 with 0):
        // one k per thread (lanes -> distinct banks), 32 n's each
        {
            const int kk = tid & 31;
            const int n0 = (tid >> 5) * 32;
            const float* src = gw + (size_t)(k0 + kk) * GATES;
            #pragma unroll
            for (int j = 0; j < 32; ++j) {
                const int n = n0 + j;
                const float v = (n < GATES) ? src[n] : 0.0f;
                Bg[n * KP + kk] = to_bf16(v);
            }
        }

        __syncthreads();

        Frag16 a;
        {
            const int kb = half * 8;
            a.i[0] = *(const int4*)&Ag[laneM * KP + kb];
            a.i[1] = *(const int4*)&Ag[laneM * KP + 16 + kb];
        }

        #pragma unroll
        for (int j = 0; j < 2; ++j) {
            const int nt = wave * 2 + j;
            Frag16 b;
            const int n = nt * 16 + laneM;
            b.i[0] = *(const int4*)&Bg[n * KP + half * 8];
            b.i[1] = *(const int4*)&Bg[n * KP + 16 + half * 8];
            acc[j] = __builtin_amdgcn_wmma_f32_16x16x32_bf16(
                false, a.v, false, b.v, (short)0, acc[j], false, false);
        }
    }

    // sigmoid gates -> LDS
    #pragma unroll
    for (int j = 0; j < 2; ++j) {
        const int col  = (wave * 2 + j) * 16 + laneM;
        const float bias = (col < GATES) ? gb[col] : 0.0f;
        #pragma unroll
        for (int r = 0; r < 8; ++r) {
            const int row = half * 8 + r;
            gbuf[row * 256 + col] = fast_sigmoid(acc[j][r] + bias);
        }
    }
    __syncthreads();

    // one thread per leaf; 16 consecutive batch entries -> coalesced store
    const int l = tid;  // 0..255
    for (int row = 0; row < BMG; ++row) {
        float p = 1.0f;
        #pragma unroll
        for (int d = 0; d < TREE_D; ++d) {
            const int node = ((1 << d) - 1) + (l >> (TREE_D - d));
            const float g  = gbuf[row * 256 + node];
            const int bit  = (l >> (TREE_D - 1 - d)) & 1;
            p *= bit ? (1.0f - g) : g;
        }
        leaf_out[(size_t)l * BATCH + bRow0 + row] = p;
    }
}

// ---------------------------------------------------------------------------
extern "C" void kernel_launch(void* const* d_in, const int* in_sizes, int n_in,
                              void* d_out, int out_size, void* d_ws, size_t ws_size,
                              hipStream_t stream) {
    const float* x  = (const float*)d_in[0];
    const float* gw = (const float*)d_in[1];
    const float* gb = (const float*)d_in[2];
    const float* pw = (const float*)d_in[3];
    const float* pb = (const float*)d_in[4];

    float* out       = (float*)d_out;
    float* leaf_out  = out + (size_t)LEAVES * BATCH * OUT_F;

    dim3 gridP(BATCH / BM, LEAVES / LT);   // 64 x 64
    softtree_proj_kernel<<<gridP, NTHREADS, 0, stream>>>(x, pw, pb, out);

    softtree_gate_kernel<<<BATCH / BMG, 256, 0, stream>>>(x, gw, gb, leaf_out);
}
